// Transformer_new_87101936763057
// MI455X (gfx1250) — compile-verified, matrix operand order checked by reading, not run on hardware
//
#include <hip/hip_runtime.h>
#include <hip/hip_bf16.h>

// Point-Transformer attention block, fused for MI455X (gfx1250, wave32).
// Dominant GEMMs run on v_wmma_f32_16x16x32_bf16 with f32 accumulation.
// key/values are pre-swizzled to bf16 WMMA B-tile images and pulled into LDS
// with global_load_async_to_lds_b128 (ASYNCcnt); weights stream from L2 with
// global_prefetch (WGP scope). All intermediates (x, h-chunks, attn logits)
// live in LDS (320KB/WGP).

#define DEV_INLINE __device__ __forceinline__

typedef __attribute__((ext_vector_type(16))) __bf16 v16bf;
typedef __attribute__((ext_vector_type(8)))  float  v8f;

static constexpr int   BB   = 4;
static constexpr int   CIN  = 128;
static constexpr int   NPT  = 2048;
static constexpr int   KN   = 16;
static constexpr int   DIM  = 256;
static constexpr int   PH   = 64;
static constexpr int   AH   = 1024;
static constexpr float EPSV = 1e-5f;

static constexpr int TP   = 8;           // points per workgroup in fused kernel
static constexpr int COLS = TP * KN;     // 128 columns per workgroup
static constexpr int NWG_MAIN = BB * NPT / TP;  // 1024
static constexpr int KV_TILE_BYTES = 2 * CIN * COLS * 2;  // 64KB per WG (key+val)

// ---------------------------------------------------------------------------
// WMMA 16x16x32 bf16 tile helpers.
// A 16x32 (MxK) / B 32x16 (KxN) lane layout (CDNA5 ISA 7.12.2):
//   lane<16 : row/col = lane,    slots 0..7 -> K 0..7,  slots 8..15 -> K 16..23
//   lane>=16: row/col = lane-16, slots 0..7 -> K 8..15, slots 8..15 -> K 24..31
// Tiles stored as 512 contiguous bf16 = [lane][16 slots]; each lane reads its
// 32 bytes with two b128 loads.
// ---------------------------------------------------------------------------
DEV_INLINE int ab_slot(int kk, int rc) {
  int lane, s;
  if      (kk <  8) { lane = rc;      s = kk;      }
  else if (kk < 16) { lane = rc + 16; s = kk - 8;  }
  else if (kk < 24) { lane = rc;      s = kk - 8;  }
  else              { lane = rc + 16; s = kk - 16; }
  return lane * 16 + s;
}

DEV_INLINE v16bf load_tile(const __bf16* tile, int lane) {
  union { uint4 u[2]; v16bf v; } t;
  const uint4* p = (const uint4*)(tile + lane * 16);
  t.u[0] = p[0];
  t.u[1] = p[1];
  return t.v;
}

DEV_INLINE v8f wmma_bf16(v16bf a, v16bf b, v8f c) {
  return __builtin_amdgcn_wmma_f32_16x16x32_bf16(false, a, false, b, (short)0,
                                                 c, false, false);
}

// Prefetch into nearest (WGP) cache level: locality 3 = keep close.
DEV_INLINE void prefetch_near(const void* p) {
  __builtin_prefetch(p, 0, 3);
}

// Async global->LDS copy (gfx1250): VDST = LDS byte offset (per-lane VGPR),
// VADDR = 64-bit global address. Tracked with ASYNCcnt.
DEV_INLINE void async_load_b128(unsigned lds_off, const void* gaddr) {
  asm volatile("global_load_async_to_lds_b128 %0, %1, off"
               :: "v"(lds_off), "v"(gaddr)
               : "memory");
}
DEV_INLINE void wait_async0() {
  asm volatile("s_wait_asynccnt 0x0" ::: "memory");
}

// ---------------------------------------------------------------------------
// BN fold: scale = g/sqrt(var+eps); bias' = b1*scale + (beta - mean*scale)
// ---------------------------------------------------------------------------
__global__ void fold_bn_kernel(const float* __restrict__ g,
                               const float* __restrict__ beta,
                               const float* __restrict__ mean,
                               const float* __restrict__ var,
                               const float* __restrict__ bias,
                               float* __restrict__ scale_out,
                               float* __restrict__ bias_out, int n) {
  int i = blockIdx.x * 256 + threadIdx.x;
  if (i < n) {
    float sc = g[i] * rsqrtf(var[i] + EPSV);
    scale_out[i] = sc;
    bias_out[i]  = bias[i] * sc + (beta[i] - mean[i] * sc);
  }
}

// ---------------------------------------------------------------------------
// Weight prep: f32 [R][C] row-major -> bf16 WMMA A-tiles [R/16][C/32][512],
// optionally row-scaled (for BN folding into Wa1).
// ---------------------------------------------------------------------------
__global__ void prep_w_kernel(const float* __restrict__ W,
                              const float* __restrict__ scale,
                              __bf16* __restrict__ out, int R, int C) {
  int i = blockIdx.x * 256 + threadIdx.x;
  if (i >= R * C) return;
  int m = i / C, kk = i % C;
  float w = W[i];
  if (scale) w *= scale[m];
  int nkc = C >> 5;
  out[(((m >> 4) * nkc) + (kk >> 5)) * 512 + ab_slot(kk & 31, m & 15)] = (__bf16)w;
}

// ---------------------------------------------------------------------------
// key/values prep: f32 (B,CIN,N,K) -> per-WG bf16 B-tile images, laid out so
// the fused kernel's LDS fill is a flat 64KB async copy:
//   kvp[wg][0:16384]   = keyB tiles   (4 kc x 8 ct x 512)
//   kvp[wg][16384:...] = valuesB tiles
// ---------------------------------------------------------------------------
__global__ void prep_kv_kernel(const float* __restrict__ key,
                               const float* __restrict__ values,
                               __bf16* __restrict__ kvp) {
  const long per = (long)CIN * COLS;                       // 16384
  long i = (long)blockIdx.x * 256 + threadIdx.x;
  if (i >= (long)NWG_MAIN * 2 * per) return;
  int  wg = (int)(i / (2 * per));
  long r  = i - (long)wg * 2 * per;
  int  t  = (int)(r / per);
  int  e  = (int)(r - (long)t * per);
  int  ch = e >> 7, col = e & 127;
  int  b  = wg >> 8, n0 = (wg & 255) * TP;
  const float* src = t ? values : key;
  float v = src[(long)b * CIN * NPT * KN + (long)ch * NPT * KN +
                (long)n0 * KN + col];
  int addr = ((ch >> 5) * 8 + (col >> 4)) * 512 + ab_slot(ch & 31, col & 15);
  kvp[(long)wg * 32768 + (long)t * 16384 + addr] = (__bf16)v;
}

// ---------------------------------------------------------------------------
// q/pe precompute: q depends only on (b,n); pe = Wp2@relu(bn(Wp1@q)).
// <1% of the FLOPs -> plain VALU. One block handles 32 points serially.
// ---------------------------------------------------------------------------
__global__ __launch_bounds__(256)
void qpe_kernel(const float* __restrict__ pos,
                const float* __restrict__ Wq, const float* __restrict__ bq,
                const float* __restrict__ Wp1, const float* __restrict__ bp1,
                const float* __restrict__ pg, const float* __restrict__ pb,
                const float* __restrict__ pm, const float* __restrict__ pv,
                const float* __restrict__ Wp2, const float* __restrict__ bp2,
                float* __restrict__ q_ws, float* __restrict__ pe_ws) {
  __shared__ float qs[DIM];
  __shared__ float us[PH];
  const int tid = threadIdx.x;
  for (int it = 0; it < 32; ++it) {
    long pt = (long)blockIdx.x * 32 + it;          // 0..8191
    int b = (int)(pt >> 11), n = (int)(pt & 2047);
    float p0 = pos[((long)b * 3 + 0) * NPT + n];
    float p1 = pos[((long)b * 3 + 1) * NPT + n];
    float p2 = pos[((long)b * 3 + 2) * NPT + n];
    float q = Wq[tid * 3 + 0] * p0 + Wq[tid * 3 + 1] * p1 +
              Wq[tid * 3 + 2] * p2 + bq[tid];
    qs[tid] = q;
    q_ws[((long)b * DIM + tid) * NPT + n] = q;
    __syncthreads();
    if (tid < PH) {
      float a = bp1[tid];
      for (int c = 0; c < DIM; ++c) a += Wp1[tid * DIM + c] * qs[c];
      float sc = pg[tid] * rsqrtf(pv[tid] + EPSV);
      a = a * sc + (pb[tid] - pm[tid] * sc);
      us[tid] = a > 0.f ? a : 0.f;
    }
    __syncthreads();
    float pe = bp2[tid];
    for (int c = 0; c < PH; ++c) pe += Wp2[tid * PH + c] * us[c];
    pe_ws[((long)b * DIM + tid) * NPT + n] = pe;
    __syncthreads();
  }
}

// ---------------------------------------------------------------------------
// Fused main kernel: per WG (8 waves, 8 points = 128 cols):
//  async-fill keyB/valB -> kf=Wk@key, v=Wv@values
//  -> x=(q-kf)*pe+pe (bf16 B-tiles), vpe=v+pe
//  h chunks (256 rows) = relu(Wa1'@x + ba1') -> attn += Wa2@h  (4 chunks)
//  softmax over k, agg = sum_k attn*vpe -> agg_ws
// LDS: bufA 64K (keyB+valB, reused as hB) | xB 64K | vpe 64K | attnAcc 128K
// ---------------------------------------------------------------------------
__global__ __launch_bounds__(256)
void fused_attn_kernel(const __bf16* __restrict__ kvp,
                       const float* __restrict__ q_ws,
                       const float* __restrict__ pe_ws,
                       const __bf16* __restrict__ Wkp, const float* __restrict__ bk,
                       const __bf16* __restrict__ Wvp, const float* __restrict__ bv,
                       const __bf16* __restrict__ Wa1p, const float* __restrict__ ba1f,
                       const __bf16* __restrict__ Wa2p, const float* __restrict__ ba2,
                       float* __restrict__ agg_ws) {
  extern __shared__ char smem[];
  __bf16* bufA    = (__bf16*)smem;              // 64KB: keyB(32K)+valB(32K) -> hB
  __bf16* xB      = (__bf16*)(smem + 65536);    // 64KB: x as B-tiles
  __bf16* vpeP    = (__bf16*)(smem + 131072);   // 64KB: v+pe, plain [r][col]
  float*  attnAcc = (float*)(smem + 196608);    // 128KB: attn logits f32

  const int tid  = threadIdx.x;
  const int lane = tid & 31;
  const int wave = tid >> 5;                    // 0..7
  const int wg   = blockIdx.x;                  // 0..1023
  const int b    = wg >> 8;
  const int n0   = (wg & 255) * TP;
  const int half8 = (lane < 16) ? 0 : 8;
  const int cn    = lane & 15;

  // ---- async LDS fill: flat 64KB copy of the pre-swizzled key/val tiles ----
  {
    const char* src = (const char*)kvp + (size_t)wg * KV_TILE_BYTES;
    unsigned base = (unsigned)(tid * 16);
#pragma unroll
    for (int i = 0; i < KV_TILE_BYTES / (256 * 16); ++i) {   // 16 iters
      unsigned off = base + (unsigned)i * 4096u;
      async_load_b128(off, src + off);
    }
  }
  wait_async0();
  __syncthreads();

  // ---- GEMM1: kf and v; epilogues write xB (B-tiles) and vpeP ----
  for (int g = 0; g < 2; ++g) {
    const __bf16* Wp   = g ? Wvp : Wkp;
    const float*  bias = g ? bv : bk;
    const __bf16* Bsrc = g ? (bufA + 16384) : bufA;
    for (int rr = 0; rr < 2; ++rr) {
      const int rt = wave * 2 + rr;            // 0..15
      v8f acc[8];
#pragma unroll
      for (int ct = 0; ct < 8; ++ct)
#pragma unroll
        for (int j = 0; j < 8; ++j) acc[ct][j] = bias[rt * 16 + j + half8];
      for (int kc = 0; kc < 4; ++kc) {
        if (kc < 3) prefetch_near(Wp + (rt * 4 + kc + 1) * 512);
        v16bf a = load_tile(Wp + (rt * 4 + kc) * 512, lane);
#pragma unroll
        for (int ct = 0; ct < 8; ++ct) {
          v16bf bm = load_tile(Bsrc + (kc * 8 + ct) * 512, lane);
          acc[ct] = wmma_bf16(a, bm, acc[ct]);
        }
      }
#pragma unroll
      for (int ct = 0; ct < 8; ++ct) {         // point p == ct
#pragma unroll
        for (int j = 0; j < 8; ++j) {
          int r = rt * 16 + j + half8;
          long qi = ((long)b * DIM + r) * NPT + n0 + ct;
          float pev = pe_ws[qi];
          if (g == 0) {
            float xv = (q_ws[qi] - acc[ct][j]) * pev + pev;
            xB[((r >> 5) * 8 + ct) * 512 + ab_slot(r & 31, cn)] = (__bf16)xv;
          } else {
            vpeP[r * COLS + ct * 16 + cn] = (__bf16)(acc[ct][j] + pev);
          }
        }
      }
    }
  }
  // init attn logits with ba2
  for (int i = tid; i < DIM * COLS; i += 256) attnAcc[i] = ba2[i >> 7];
  __syncthreads();

  // ---- hidden chunks: h = relu(Wa1'@x + ba1'), attn += Wa2@h ----
  for (int chunk = 0; chunk < 4; ++chunk) {
    for (int rr = 0; rr < 2; ++rr) {           // GEMM2 -> hB (bufA)
      const int lrt = wave + rr * 8;           // 0..15 in chunk
      const int grt = chunk * 16 + lrt;        // 0..63 global AH tile
      v8f acc[8];
#pragma unroll
      for (int ct = 0; ct < 8; ++ct)
#pragma unroll
        for (int j = 0; j < 8; ++j) acc[ct][j] = 0.f;
      for (int kc = 0; kc < 8; ++kc) {
        if (kc < 7) prefetch_near(Wa1p + (grt * 8 + kc + 1) * 512);
        v16bf a = load_tile(Wa1p + (grt * 8 + kc) * 512, lane);
#pragma unroll
        for (int ct = 0; ct < 8; ++ct) {
          v16bf bm = load_tile(xB + (kc * 8 + ct) * 512, lane);
          acc[ct] = wmma_bf16(a, bm, acc[ct]);
        }
      }
#pragma unroll
      for (int ct = 0; ct < 8; ++ct)
#pragma unroll
        for (int j = 0; j < 8; ++j) {
          int lr = lrt * 16 + j + half8;       // 0..255 within chunk
          float h = acc[ct][j] + ba1f[chunk * 256 + lr];
          h = h > 0.f ? h : 0.f;
          bufA[((lr >> 5) * 8 + ct) * 512 + ab_slot(lr & 31, cn)] = (__bf16)h;
        }
    }
    __syncthreads();
    for (int rr = 0; rr < 2; ++rr) {           // GEMM3 accumulate attn
      const int rt = wave + rr * 8;            // 0..15
      v8f acc[8];
#pragma unroll
      for (int ct = 0; ct < 8; ++ct)
#pragma unroll
        for (int j = 0; j < 8; ++j) acc[ct][j] = 0.f;
      for (int kc = 0; kc < 8; ++kc) {
        if (kc < 7)
          prefetch_near(Wa2p + (rt * 32 + chunk * 8 + kc + 1) * 512);
        v16bf a = load_tile(Wa2p + (rt * 32 + chunk * 8 + kc) * 512, lane);
#pragma unroll
        for (int ct = 0; ct < 8; ++ct) {
          v16bf bm = load_tile(bufA + (kc * 8 + ct) * 512, lane);
          acc[ct] = wmma_bf16(a, bm, acc[ct]);
        }
      }
#pragma unroll
      for (int ct = 0; ct < 8; ++ct)
#pragma unroll
        for (int j = 0; j < 8; ++j) {
          int r = rt * 16 + j + half8;
          attnAcc[r * COLS + ct * 16 + cn] += acc[ct][j];
        }
    }
    __syncthreads();
  }

  // ---- softmax over k (16) and weighted reduce ----
  for (int idx = tid; idx < DIM * TP; idx += 256) {
    int r = idx >> 3, p = idx & 7;
    const float* row = attnAcc + r * COLS + p * 16;
    float m = row[0];
#pragma unroll
    for (int k = 1; k < 16; ++k) m = fmaxf(m, row[k]);
    float e[16], s = 0.f;
#pragma unroll
    for (int k = 0; k < 16; ++k) { e[k] = __expf(row[k] - m); s += e[k]; }
    float inv = 1.f / s, a = 0.f;
#pragma unroll
    for (int k = 0; k < 16; ++k)
      a += e[k] * inv * (float)vpeP[r * COLS + p * 16 + k];
    agg_ws[((long)b * DIM + r) * NPT + n0 + p] = a;
  }
}

// ---------------------------------------------------------------------------
// Final projection: out = We @ agg + be   ([256x256] x [256 x 8192], WMMA)
// One WG handles 128 points (8 col tiles); 64 WGs.
// ---------------------------------------------------------------------------
__global__ __launch_bounds__(256)
void out_proj_kernel(const float* __restrict__ agg_ws,
                     const __bf16* __restrict__ Wep,
                     const float* __restrict__ be,
                     float* __restrict__ out) {
  extern __shared__ char smem[];
  __bf16* aggB = (__bf16*)smem;                // 64KB B-tiles
  const int tid  = threadIdx.x;
  const int lane = tid & 31;
  const int wave = tid >> 5;
  const int wg   = blockIdx.x;                 // 0..63
  const int b    = wg >> 4;
  const int nc0  = (wg & 15) * 128;
  const int half8 = (lane < 16) ? 0 : 8;
  const int cn    = lane & 15;

  for (int i = tid; i < DIM * 128; i += 256) {
    int r = i >> 7, col = i & 127;
    float v = agg_ws[((long)b * DIM + r) * NPT + nc0 + col];
    aggB[((r >> 5) * 8 + (col >> 4)) * 512 + ab_slot(r & 31, col & 15)] = (__bf16)v;
  }
  __syncthreads();

  for (int rr = 0; rr < 2; ++rr) {
    const int rt = wave + rr * 8;              // 0..15
    v8f acc[8];
#pragma unroll
    for (int ct = 0; ct < 8; ++ct)
#pragma unroll
      for (int j = 0; j < 8; ++j) acc[ct][j] = be[rt * 16 + j + half8];
    for (int kc = 0; kc < 8; ++kc) {
      if (kc < 7) prefetch_near(Wep + (rt * 8 + kc + 1) * 512);
      v16bf a = load_tile(Wep + (rt * 8 + kc) * 512, lane);
#pragma unroll
      for (int ct = 0; ct < 8; ++ct) {
        v16bf bm = load_tile(aggB + (kc * 8 + ct) * 512, lane);
        acc[ct] = wmma_bf16(a, bm, acc[ct]);
      }
    }
#pragma unroll
    for (int ct = 0; ct < 8; ++ct)
#pragma unroll
      for (int j = 0; j < 8; ++j) {
        int r = rt * 16 + j + half8;
        out[((long)b * DIM + r) * NPT + nc0 + ct * 16 + cn] = acc[ct][j];
      }
  }
}

// ---------------------------------------------------------------------------
extern "C" void kernel_launch(void* const* d_in, const int* in_sizes, int n_in,
                              void* d_out, int out_size, void* d_ws, size_t ws_size,
                              hipStream_t stream) {
  const float* key    = (const float*)d_in[0];
  const float* values = (const float*)d_in[1];
  const float* pos    = (const float*)d_in[2];
  const float* Wk  = (const float*)d_in[3];
  const float* bk  = (const float*)d_in[4];
  const float* Wq  = (const float*)d_in[5];
  const float* bq  = (const float*)d_in[6];
  const float* Wv  = (const float*)d_in[7];
  const float* bv  = (const float*)d_in[8];
  const float* Wp1 = (const float*)d_in[9];
  const float* bp1 = (const float*)d_in[10];
  const float* pg  = (const float*)d_in[11];
  const float* pb  = (const float*)d_in[12];
  const float* pm  = (const float*)d_in[13];
  const float* pv  = (const float*)d_in[14];
  const float* Wp2 = (const float*)d_in[15];
  const float* bp2 = (const float*)d_in[16];
  const float* Wa1 = (const float*)d_in[17];
  const float* ba1 = (const float*)d_in[18];
  const float* ag  = (const float*)d_in[19];
  const float* abt = (const float*)d_in[20];
  const float* am  = (const float*)d_in[21];
  const float* av  = (const float*)d_in[22];
  const float* Wa2 = (const float*)d_in[23];
  const float* ba2 = (const float*)d_in[24];
  const float* We  = (const float*)d_in[25];
  const float* be  = (const float*)d_in[26];
  float* out = (float*)d_out;

  char* ws = (char*)d_ws;
  size_t off = 0;
  auto wsalloc = [&](size_t bytes) -> char* {
    char* p = ws + off;
    off += (bytes + 255) & ~(size_t)255;
    return p;
  };
  float*  q_ws   = (float*)wsalloc(sizeof(float) * BB * DIM * NPT);   // 8MB
  float*  pe_ws  = (float*)wsalloc(sizeof(float) * BB * DIM * NPT);   // 8MB
  float*  agg_ws = (float*)wsalloc(sizeof(float) * BB * DIM * NPT);   // 8MB
  __bf16* kvp    = (__bf16*)wsalloc((size_t)NWG_MAIN * KV_TILE_BYTES); // 64MB
  __bf16* Wkp    = (__bf16*)wsalloc(2 * DIM * CIN);
  __bf16* Wvp    = (__bf16*)wsalloc(2 * DIM * CIN);
  __bf16* Wa1p   = (__bf16*)wsalloc(2 * AH * DIM);
  __bf16* Wa2p   = (__bf16*)wsalloc(2 * DIM * AH);
  __bf16* Wep    = (__bf16*)wsalloc(2 * DIM * DIM);
  float*  scaleA = (float*)wsalloc(sizeof(float) * AH);
  float*  ba1f   = (float*)wsalloc(sizeof(float) * AH);
  (void)ws_size; (void)in_sizes; (void)n_in; (void)out_size;

  fold_bn_kernel<<<(AH + 255) / 256, 256, 0, stream>>>(ag, abt, am, av, ba1,
                                                       scaleA, ba1f, AH);
  prep_w_kernel<<<(DIM * CIN + 255) / 256, 256, 0, stream>>>(Wk, nullptr, Wkp, DIM, CIN);
  prep_w_kernel<<<(DIM * CIN + 255) / 256, 256, 0, stream>>>(Wv, nullptr, Wvp, DIM, CIN);
  prep_w_kernel<<<(AH * DIM + 255) / 256, 256, 0, stream>>>(Wa1, scaleA, Wa1p, AH, DIM);
  prep_w_kernel<<<(DIM * AH + 255) / 256, 256, 0, stream>>>(Wa2, nullptr, Wa2p, DIM, AH);
  prep_w_kernel<<<(DIM * DIM + 255) / 256, 256, 0, stream>>>(We, nullptr, Wep, DIM, DIM);
  {
    long total = (long)NWG_MAIN * 2 * CIN * COLS;          // 33.5M elements
    prep_kv_kernel<<<(int)((total + 255) / 256), 256, 0, stream>>>(key, values, kvp);
  }
  qpe_kernel<<<256, 256, 0, stream>>>(pos, Wq, bq, Wp1, bp1, pg, pb, pm, pv,
                                      Wp2, bp2, q_ws, pe_ws);
  fused_attn_kernel<<<NWG_MAIN, 256, 327680, stream>>>(
      kvp, q_ws, pe_ws, Wkp, bk, Wvp, bv, Wa1p, ba1f, Wa2p, ba2, agg_ws);
  out_proj_kernel<<<64, 256, 65536, stream>>>(agg_ws, Wep, be, out);
}
